// SimpleSelectiveSSM_83502754169501
// MI455X (gfx1250) — compile-verified, matrix operand order checked by reading, not run on hardware
//
#include <hip/hip_runtime.h>
#include <math.h>

// ---------------- problem constants (from reference) ----------------
#define NBATCH  2
#define SEQLEN  2048
#define DM      1024            // d_model (= GEMM K)
#define DSTATE  16
#define PROJ    (2*DM + 2*DM*DSTATE)   // 34816 = GEMM N

// ---------------- GEMM tiling ----------------
#define BLK_M   128
#define BLK_N   128
#define KTILE   32              // bf16 WMMA K

typedef __attribute__((ext_vector_type(16))) __bf16 v16bf;
typedef __attribute__((ext_vector_type(8)))  float  v8f;
typedef __attribute__((ext_vector_type(4)))  int    v4i;

// ------------- async global->LDS support (gfx1250 ASYNCcnt path) -------------
#if defined(__has_builtin)
#if __has_builtin(__builtin_amdgcn_global_load_async_to_lds_b128)
#define HAVE_ASYNC_LDS 1
#endif
#endif
#ifndef HAVE_ASYNC_LDS
#define HAVE_ASYNC_LDS 0
#endif

#if defined(__has_builtin)
#if __has_builtin(__builtin_amdgcn_s_wait_asynccnt)
#define WAIT_ASYNC(n) __builtin_amdgcn_s_wait_asynccnt(n)
#endif
#endif
#ifndef WAIT_ASYNC
#if HAVE_ASYNC_LDS
#define WAIT_ASYNC(n) asm volatile("s_wait_asynccnt %0" ::"i"(n) : "memory")
#else
#define WAIT_ASYNC(n)
#endif
#endif

__device__ __forceinline__ void copy16_to_lds(const void* g, void* l) {
#if HAVE_ASYNC_LDS
  __builtin_amdgcn_global_load_async_to_lds_b128(
      (__attribute__((address_space(1))) v4i*)g,
      (__attribute__((address_space(3))) v4i*)l, 0, 0);
#else
  *(ulonglong2*)l = *(const ulonglong2*)g;
#endif
}

union FragU {
  v16bf v;
  unsigned long long q[4];
};

// Load one 16x32 bf16 operand fragment from an LDS tile stored row-major
// (32 bf16 per row).  Layout follows CDNA5 ISA 7.12.2 (16-bit A/B matrix):
// lane&15 selects the row, lane>>4 selects the K-half:
//   elems 0..7  = K (lane<16 ? 0..7  : 8..15)
//   elems 8..15 = K (lane<16 ? 16..23: 24..31)
__device__ __forceinline__ v16bf load_frag(const __bf16* tile, int lane) {
  const int row = lane & 15;
  const int hi  = (lane >> 4) & 1;
  const __bf16* r = tile + row * KTILE;
  FragU u;
  const unsigned long long* p0 = (const unsigned long long*)(r + hi * 8);
  u.q[0] = p0[0];
  u.q[1] = p0[1];
  const unsigned long long* p1 = (const unsigned long long*)(r + 16 + hi * 8);
  u.q[2] = p1[0];
  u.q[3] = p1[1];
  return u.v;
}

// -------------------------------------------------------------------
// Pre-pack fp32 -> bf16 hi/lo planes (fp32-accurate split: v = hi + lo)
// -------------------------------------------------------------------
__global__ __launch_bounds__(256)
void pack_planes(const float* __restrict__ src, __bf16* __restrict__ hi,
                 __bf16* __restrict__ lo, size_t n) {
  const size_t i = (size_t)blockIdx.x * 256 + threadIdx.x;
  if (i < n) {
    const float v = src[i];
    const __bf16 h = (__bf16)v;
    hi[i] = h;
    lo[i] = (__bf16)(v - (float)h);
  }
}

// -------------------------------------------------------------------
// Phase 1 (packed path): xp[m,p] = sum_k x[m,k]*W[p,k] + b[p] using
// pre-split bf16 planes, async double-buffered LDS staging, bf16x3 WMMA.
// -------------------------------------------------------------------
__global__ __launch_bounds__(256)
void ssm_proj_gemm_packed(const __bf16* __restrict__ Xhi, const __bf16* __restrict__ Xlo,
                          const __bf16* __restrict__ Whi, const __bf16* __restrict__ Wlo,
                          const float* __restrict__ bias, float* __restrict__ xp,
                          int s0, int CH)
{
  const int M_total = NBATCH * CH;
  const int block_m = blockIdx.x * BLK_M;
  const int block_n = blockIdx.y * BLK_N;
  const int tid  = threadIdx.x;
  const int wave = tid >> 5;
  const int lane = tid & 31;
  const int wm   = wave >> 2;   // 0..1 -> M offset 64*wm
  const int wn   = wave & 3;    // 0..3 -> N offset 32*wn

  // double-buffered staging: 2 bufs x 4 planes x 8KB = 64KB LDS
  __shared__ __attribute__((aligned(16))) __bf16 sAhi[2][BLK_M * KTILE];
  __shared__ __attribute__((aligned(16))) __bf16 sAlo[2][BLK_M * KTILE];
  __shared__ __attribute__((aligned(16))) __bf16 sBhi[2][BLK_N * KTILE];
  __shared__ __attribute__((aligned(16))) __bf16 sBlo[2][BLK_N * KTILE];

  // per-thread staging addresses: thread t covers 32 bytes of each 8KB tile
  const int r     = tid >> 1;          // tile row (0..127)
  const int halfb = (tid & 1) * 32;    // byte offset within 64B row
  int m = block_m + r;
  if (m >= M_total) m = M_total - 1;   // clamp: garbage rows are masked in epilogue
  const size_t xrow = (size_t)((m / CH) * SEQLEN + s0 + (m % CH));
  const size_t prow = (size_t)(block_n + r);

  const char* gAhi = (const char*)(Xhi + xrow * DM) + halfb;
  const char* gAlo = (const char*)(Xlo + xrow * DM) + halfb;
  const char* gBhi = (const char*)(Whi + prow * DM) + halfb;
  const char* gBlo = (const char*)(Wlo + prow * DM) + halfb;
  const int ldsoff = tid * 32;         // byte offset in staging buffer

#define STAGE(buf, k0)                                                        \
  do {                                                                        \
    const int kb = (k0) * 2; /* byte offset along K */                        \
    copy16_to_lds(gAhi + kb,      (char*)sAhi[buf] + ldsoff);                 \
    copy16_to_lds(gAhi + kb + 16, (char*)sAhi[buf] + ldsoff + 16);            \
    copy16_to_lds(gAlo + kb,      (char*)sAlo[buf] + ldsoff);                 \
    copy16_to_lds(gAlo + kb + 16, (char*)sAlo[buf] + ldsoff + 16);            \
    copy16_to_lds(gBhi + kb,      (char*)sBhi[buf] + ldsoff);                 \
    copy16_to_lds(gBhi + kb + 16, (char*)sBhi[buf] + ldsoff + 16);            \
    copy16_to_lds(gBlo + kb,      (char*)sBlo[buf] + ldsoff);                 \
    copy16_to_lds(gBlo + kb + 16, (char*)sBlo[buf] + ldsoff + 16);            \
  } while (0)

  v8f acc[4][2];
#pragma unroll
  for (int i = 0; i < 4; ++i)
#pragma unroll
    for (int j = 0; j < 2; ++j)
      acc[i][j] = (v8f)(0.0f);

  STAGE(0, 0);

  const int NK = DM / KTILE;
  for (int it = 0; it < NK; ++it) {
    const int cur = it & 1;
    if (it + 1 < NK) {
      STAGE(cur ^ 1, (it + 1) * KTILE);
      WAIT_ASYNC(8);      // previous stage's 8 async ops retired; next 8 in flight
    } else {
      WAIT_ASYNC(0);
    }
    __syncthreads();

    v16bf ah[4], al[4], bh[2], bl[2];
#pragma unroll
    for (int mt = 0; mt < 4; ++mt) {
      const int base = (wm * 64 + mt * 16) * KTILE;
      ah[mt] = load_frag(sAhi[cur] + base, lane);
      al[mt] = load_frag(sAlo[cur] + base, lane);
    }
#pragma unroll
    for (int nt = 0; nt < 2; ++nt) {
      const int base = (wn * 32 + nt * 16) * KTILE;
      bh[nt] = load_frag(sBhi[cur] + base, lane);
      bl[nt] = load_frag(sBlo[cur] + base, lane);
    }

#pragma unroll
    for (int mt = 0; mt < 4; ++mt)
#pragma unroll
      for (int nt = 0; nt < 2; ++nt) {
        acc[mt][nt] = __builtin_amdgcn_wmma_f32_16x16x32_bf16(
            false, ah[mt], false, bh[nt], (short)0, acc[mt][nt], false, false);
        acc[mt][nt] = __builtin_amdgcn_wmma_f32_16x16x32_bf16(
            false, al[mt], false, bh[nt], (short)0, acc[mt][nt], false, false);
        acc[mt][nt] = __builtin_amdgcn_wmma_f32_16x16x32_bf16(
            false, ah[mt], false, bl[nt], (short)0, acc[mt][nt], false, false);
      }
    __syncthreads();   // all reads of buf[cur] done before it is restaged
  }
#undef STAGE

  // epilogue: C layout (ISA 7.12.2): lane<16 -> M=r, lane>=16 -> M=r+8, N=lane&15
  const int lrow = ((lane >> 4) & 1) << 3;
  const int lcol = lane & 15;
#pragma unroll
  for (int mt = 0; mt < 4; ++mt)
#pragma unroll
    for (int nt = 0; nt < 2; ++nt) {
      const int n = block_n + wn * 32 + nt * 16 + lcol;
      const float bv = bias[n];
#pragma unroll
      for (int rr = 0; rr < 8; ++rr) {
        const int mm = block_m + wm * 64 + mt * 16 + lrow + rr;
        if (mm < M_total)
          xp[(size_t)mm * PROJ + n] = acc[mt][nt][rr] + bv;
      }
    }
}

// -------------------------------------------------------------------
// Phase 1 (fallback path, ws too small for packed planes): converts
// fp32 -> bf16 hi/lo in-kernel while staging to LDS.
// -------------------------------------------------------------------
__global__ __launch_bounds__(256)
void ssm_proj_gemm_cvt(const float* __restrict__ x, const float* __restrict__ W,
                       const float* __restrict__ bias, float* __restrict__ xp,
                       int s0, int CH)
{
  const int M_total = NBATCH * CH;
  const int block_m = blockIdx.x * BLK_M;
  const int block_n = blockIdx.y * BLK_N;
  const int tid  = threadIdx.x;
  const int wave = tid >> 5;
  const int lane = tid & 31;
  const int wm   = wave >> 2;
  const int wn   = wave & 3;

  __shared__ __bf16 sAhi[BLK_M * KTILE];
  __shared__ __bf16 sAlo[BLK_M * KTILE];
  __shared__ __bf16 sBhi[BLK_N * KTILE];
  __shared__ __bf16 sBlo[BLK_N * KTILE];

  v8f acc[4][2];
#pragma unroll
  for (int i = 0; i < 4; ++i)
#pragma unroll
    for (int j = 0; j < 2; ++j)
      acc[i][j] = (v8f)(0.0f);

  for (int k0 = 0; k0 < DM; k0 += KTILE) {
#pragma unroll
    for (int i = tid; i < BLK_M * KTILE; i += 256) {
      const int r = i >> 5;
      const int c = i & 31;
      const int m = block_m + r;
      float v = 0.0f;
      if (m < M_total) {
        const int b = m / CH;
        const int s = s0 + (m % CH);
        v = x[((size_t)(b * SEQLEN + s)) * DM + k0 + c];
      }
      const __bf16 h = (__bf16)v;
      sAhi[i] = h;
      sAlo[i] = (__bf16)(v - (float)h);
    }
#pragma unroll
    for (int i = tid; i < BLK_N * KTILE; i += 256) {
      const int r = i >> 5;
      const int c = i & 31;
      const int p = block_n + r;
      const float v = W[(size_t)p * DM + k0 + c];
      const __bf16 h = (__bf16)v;
      sBhi[i] = h;
      sBlo[i] = (__bf16)(v - (float)h);
    }
    __syncthreads();

    v16bf ah[4], al[4], bh[2], bl[2];
#pragma unroll
    for (int mt = 0; mt < 4; ++mt) {
      const int base = (wm * 64 + mt * 16) * KTILE;
      ah[mt] = load_frag(sAhi + base, lane);
      al[mt] = load_frag(sAlo + base, lane);
    }
#pragma unroll
    for (int nt = 0; nt < 2; ++nt) {
      const int base = (wn * 32 + nt * 16) * KTILE;
      bh[nt] = load_frag(sBhi + base, lane);
      bl[nt] = load_frag(sBlo + base, lane);
    }
#pragma unroll
    for (int mt = 0; mt < 4; ++mt)
#pragma unroll
      for (int nt = 0; nt < 2; ++nt) {
        acc[mt][nt] = __builtin_amdgcn_wmma_f32_16x16x32_bf16(
            false, ah[mt], false, bh[nt], (short)0, acc[mt][nt], false, false);
        acc[mt][nt] = __builtin_amdgcn_wmma_f32_16x16x32_bf16(
            false, al[mt], false, bh[nt], (short)0, acc[mt][nt], false, false);
        acc[mt][nt] = __builtin_amdgcn_wmma_f32_16x16x32_bf16(
            false, ah[mt], false, bl[nt], (short)0, acc[mt][nt], false, false);
      }
    __syncthreads();
  }

  const int lrow = ((lane >> 4) & 1) << 3;
  const int lcol = lane & 15;
#pragma unroll
  for (int mt = 0; mt < 4; ++mt)
#pragma unroll
    for (int nt = 0; nt < 2; ++nt) {
      const int n = block_n + wn * 32 + nt * 16 + lcol;
      const float bv = bias[n];
#pragma unroll
      for (int rr = 0; rr < 8; ++rr) {
        const int mm = block_m + wm * 64 + mt * 16 + lrow + rr;
        if (mm < M_total)
          xp[(size_t)mm * PROJ + n] = acc[mt][nt][rr] + bv;
      }
    }
}

// -------------------------------------------------------------------
// Phase 2: sequential selective scan over one chunk.
// 16 lanes (= 16 states) per (batch, channel); h carried in workspace.
// -------------------------------------------------------------------
__global__ __launch_bounds__(256)
void ssm_scan(const float* __restrict__ xp, const float* __restrict__ A,
              const float* __restrict__ Dvec, float* __restrict__ h_ws,
              float* __restrict__ out, int s0, int CH)
{
  const int gid = blockIdx.x * (256 / DSTATE) + (threadIdx.x >> 4); // 0..2047
  const int n   = threadIdx.x & 15;
  const int b   = gid / DM;
  const int d   = gid % DM;

  const float Adn = A[d * DSTATE + n];
  const float Dd  = Dvec[d];
  float h = h_ws[(size_t)gid * DSTATE + n];

  const float* row = xp + (size_t)(b * CH) * PROJ;
  for (int t = 0; t < CH; ++t) {
    const float u  = row[d];
    const float dr = row[DM + d];
    const float Bv = row[2 * DM + d * DSTATE + n];
    const float Cv = row[2 * DM + DM * DSTATE + d * DSTATE + n];
    const float sp   = (dr > 20.0f) ? dr : log1pf(__expf(dr));  // softplus
    const float Abar = __expf(Adn * sp);
    h = Abar * h + Bv * u;
    float y = Cv * h;
#pragma unroll
    for (int o = 8; o > 0; o >>= 1) y += __shfl_xor(y, o, 16);
    if (n == 0)
      out[(size_t)(b * SEQLEN + s0 + t) * DM + d] = y + Dd * u;
    row += PROJ;
  }
  h_ws[(size_t)gid * DSTATE + n] = h;
}

__global__ void ssm_init_h(float* h_ws) {
  const int i = blockIdx.x * blockDim.x + threadIdx.x;
  if (i < NBATCH * DM * DSTATE) h_ws[i] = 0.0f;
}

// -------------------------------------------------------------------
extern "C" void kernel_launch(void* const* d_in, const int* in_sizes, int n_in,
                              void* d_out, int out_size, void* d_ws, size_t ws_size,
                              hipStream_t stream) {
  (void)in_sizes; (void)n_in; (void)out_size;
  const float* x    = (const float*)d_in[0];
  const float* W    = (const float*)d_in[1];
  const float* bias = (const float*)d_in[2];
  const float* A    = (const float*)d_in[3];
  const float* Dv   = (const float*)d_in[4];
  float* out = (float*)d_out;

  const size_t ALIGN = 256;
  const size_t nW = (size_t)PROJ * DM;              // 35.65M elems
  const size_t nX = (size_t)NBATCH * SEQLEN * DM;   // 4.19M elems

  size_t off = 0;
  auto carve = [&](size_t bytes) -> char* {
    char* p = (char*)d_ws + off;
    off = (off + bytes + ALIGN - 1) & ~(ALIGN - 1);
    return p;
  };
  float* h_ws = (float*)carve((size_t)NBATCH * DM * DSTATE * sizeof(float));
  const size_t pack_begin = off;
  char* pWhi = carve(nW * 2);
  char* pWlo = carve(nW * 2);
  char* pXhi = carve(nX * 2);
  char* pXlo = carve(nX * 2);
  const size_t pack_end = off;
  const size_t min_chunk = (size_t)NBATCH * 16 * PROJ * sizeof(float);
  const bool packed_ok = ws_size >= pack_end + min_chunk;

  ssm_init_h<<<dim3((NBATCH * DM * DSTATE + 255) / 256), dim3(256), 0, stream>>>(h_ws);

  size_t xp_off = packed_ok ? pack_end : pack_begin;
  float* xp = (float*)((char*)d_ws + xp_off);
  const size_t avail = (ws_size > xp_off) ? ws_size - xp_off : 0;
  int CH = SEQLEN;
  while (CH > 16 && (size_t)NBATCH * CH * PROJ * sizeof(float) > avail) CH >>= 1;

  if (packed_ok) {
    pack_planes<<<dim3((unsigned)((nW + 255) / 256)), dim3(256), 0, stream>>>(
        W, (__bf16*)pWhi, (__bf16*)pWlo, nW);
    pack_planes<<<dim3((unsigned)((nX + 255) / 256)), dim3(256), 0, stream>>>(
        x, (__bf16*)pXhi, (__bf16*)pXlo, nX);
  }

  for (int s0 = 0; s0 < SEQLEN; s0 += CH) {
    const int M_total = NBATCH * CH;
    dim3 grid((M_total + BLK_M - 1) / BLK_M, PROJ / BLK_N);
    if (packed_ok) {
      ssm_proj_gemm_packed<<<grid, dim3(256), 0, stream>>>(
          (const __bf16*)pXhi, (const __bf16*)pXlo,
          (const __bf16*)pWhi, (const __bf16*)pWlo, bias, xp, s0, CH);
    } else {
      ssm_proj_gemm_cvt<<<grid, dim3(256), 0, stream>>>(x, W, bias, xp, s0, CH);
    }
    ssm_scan<<<dim3((NBATCH * DM) / (256 / DSTATE)), dim3(256), 0, stream>>>(
        xp, A, Dv, h_ws, out, s0, CH);
  }
}
